// PatchRangeAttention_15135464751306
// MI455X (gfx1250) — matrix operand order repaired and checked by reading, not compiled, not run on hardware
//
#include <hip/hip_runtime.h>
#include <hip/hip_bf16.h>

// ---------------------------------------------------------------------------
// PatchRangeAttention for MI455X (gfx1250, wave32)
//   WMMA f32_16x16x32_f16 + async global->LDS staging (ASYNCcnt) pipeline.
//   B=8, N=1025 (32x32 grid + CLS), DIM=768, H=12, Dh=64, radius R=3
// ---------------------------------------------------------------------------

typedef _Float16 f16;
typedef __attribute__((ext_vector_type(16))) _Float16 v16h;
typedef __attribute__((ext_vector_type(8)))  _Float16 v8h;
typedef __attribute__((ext_vector_type(8)))  float    v8f;
typedef __attribute__((ext_vector_type(4)))  float    v4f;

union V16 { v16h v; v8h h[2]; };

#define WMMA_F16(Afr, Bfr, Cfr) \
  __builtin_amdgcn_wmma_f32_16x16x32_f16(false, (Afr), false, (Bfr), (short)0, (Cfr), false, false)

#define NEG_INF (-__builtin_inff())

// Async copy 16B global -> LDS. ldsoff = wave-relative LDS byte offset
// (low 32 bits of a generic pointer to __shared__), goff = byte offset VGPR,
// base = 64-bit SGPR base. GVS mode, tracked by ASYNCcnt.
#define ASYNC_B128(ldsoff, goff, base)                                   \
  asm volatile("global_load_async_to_lds_b128 %0, %1, %2"                \
               :: "v"(ldsoff), "v"(goff), "s"(base) : "memory")

// ws layout (bytes):
//   Q  f16 [96][1025][64]   @ 0           (12,595,200)
//   K  f16 [96][1025][64]   @ 12,595,200  (12,595,200)
//   Vt f16 [96][64][1040]   @ 25,190,400  (12,779,520)
//   AO f16 [8200][768]      @ 37,969,920  (12,607,488)  -- also reused as Xh
//   Wqkvt f16 [2304][768]   @ 50,577,408  ( 3,538,944)
//   Wpt   f16 [768][768]    @ 54,116,352  ( 1,179,648)

// ===========================================================================
// Prepass A: x (f32) -> Xh (f16), 8 elems/thread
// ===========================================================================
__global__ __launch_bounds__(256) void cvt_x_kernel(
    const float* __restrict__ x, f16* __restrict__ Xh)
{
  const size_t idx = ((size_t)blockIdx.x * 256 + threadIdx.x) * 8;
  if (idx < (size_t)8200 * 768) {
    const v4f a = *(const v4f*)(x + idx);
    const v4f b = *(const v4f*)(x + idx + 4);
    v8h o;
    o[0] = (f16)a[0]; o[1] = (f16)a[1]; o[2] = (f16)a[2]; o[3] = (f16)a[3];
    o[4] = (f16)b[0]; o[5] = (f16)b[1]; o[6] = (f16)b[2]; o[7] = (f16)b[3];
    *(v8h*)(Xh + idx) = o;
  }
}

// ===========================================================================
// Prepass B: W[K][N] (f32) -> Wt[N][K] (f16), LDS-tiled 32x32 transpose
// ===========================================================================
__global__ __launch_bounds__(256) void tconv_kernel(
    const float* __restrict__ W, f16* __restrict__ Wt, int K, int N)
{
  __shared__ f16 tile[32][33];
  const int tx = threadIdx.x & 31, ty = threadIdx.x >> 5;
  const int nbase = blockIdx.x * 32, kbase = blockIdx.y * 32;
#pragma unroll
  for (int j = 0; j < 4; ++j)
    tile[ty + 8 * j][tx] = (f16)W[(size_t)(kbase + ty + 8 * j) * N + nbase + tx];
  __syncthreads();
#pragma unroll
  for (int j = 0; j < 4; ++j)
    Wt[(size_t)(nbase + ty + 8 * j) * K + kbase + tx] = tile[tx][ty + 8 * j];
}

// ===========================================================================
// Kernel 1: qkv = Xh @ Wqkvt^T + bqkv -> scatter Q (pre-scaled), K, Vt (f16)
// Block tile 64(M) x 128(N), K-step 32, 8 waves, async double-buffered LDS.
// ===========================================================================
__global__ __launch_bounds__(256) void qkv_gemm_kernel(
    const f16* __restrict__ Xh, const f16* __restrict__ Wt,
    const float* __restrict__ bqkv,
    f16* __restrict__ Qb, f16* __restrict__ Kb, f16* __restrict__ Vt)
{
  __shared__ __align__(16) f16 As[2][64][40];   // [m][k], row padded to 40
  __shared__ __align__(16) f16 Bs[2][128][40];  // [n][k], row padded to 40

  const int tid  = threadIdx.x;
  const int lane = tid & 31;
  const int w    = tid >> 5;
  const int mw   = w & 3;          // wave M sub-tile (16 rows)
  const int nw   = w >> 2;         // wave N sub-tile (64 cols)
  const int lm   = lane & 15;
  const int hh   = lane >> 4;
  const int m0   = blockIdx.y * 64;
  const int n0   = blockIdx.x * 128;

  auto stage = [&](int kk, int buf) {
    const int k0 = kk * 32;
    {
      const int row = tid >> 2, seg = tid & 3;
      int gm = m0 + row; if (gm > 8199) gm = 8199;
      const unsigned la = (unsigned)(size_t)&As[buf][row][seg * 8];
      const int go = (gm * 768 + k0 + seg * 8) * 2;
      ASYNC_B128(la, go, Xh);
    }
#pragma unroll
    for (int it = 0; it < 2; ++it) {
      const int idx = tid + it * 256;
      const int row = idx >> 2, seg = idx & 3;
      const unsigned lb = (unsigned)(size_t)&Bs[buf][row][seg * 8];
      const int go = ((n0 + row) * 768 + k0 + seg * 8) * 2;
      ASYNC_B128(lb, go, Wt);
    }
    asm volatile("s_wait_xcnt 0x0" ::: "memory");  // addr VGPRs safe to reuse
  };

  v8f acc[4] = {};
  stage(0, 0);

  for (int kk = 0; kk < 24; ++kk) {
    const int buf = kk & 1;
    asm volatile("s_wait_asynccnt 0x0" ::: "memory");  // my tile data in LDS
    __syncthreads();                                   // everyone's tile data
    if (kk + 1 < 24) stage(kk + 1, buf ^ 1);

    V16 a;  // A frag: lane m = lm, K run {16h..16h+15}
    a.h[0] = *(const v8h*)&As[buf][mw * 16 + lm][16 * hh];
    a.h[1] = *(const v8h*)&As[buf][mw * 16 + lm][16 * hh + 8];
#pragma unroll
    for (int j = 0; j < 4; ++j) {
      const f16* bp = &Bs[buf][nw * 64 + j * 16 + lm][16 * hh];
      V16 bfr;  // B frag: lane n, K run {16h..16h+15}
      bfr.h[0] = *(const v8h*)bp;
      bfr.h[1] = *(const v8h*)(bp + 8);
      acc[j] = WMMA_F16(a.v, bfr.v, acc[j]);
    }
    __syncthreads();  // done reading buf before it is refilled next+1 round
  }

  // ---- epilogue: bias + scatter into Q / K / Vt ----
  const int mbase = m0 + mw * 16 + 8 * hh;
  const int bb = mbase / 1025;
  const int tt = mbase % 1025;
#pragma unroll
  for (int j = 0; j < 4; ++j) {
    const int n     = n0 + nw * 64 + j * 16 + lm;
    const float bias = bqkv[n];
    const int which = n / 768;      // 0=q 1=k 2=v (uniform per sub-tile)
    const int d     = n % 768;
    const int head  = d >> 6;
    const int dh    = d & 63;
#pragma unroll
    for (int r = 0; r < 8; ++r) {
      const int m = mbase + r;
      if (m < 8200) {
        int t = tt + r, b = bb;
        if (t > 1024) { t -= 1025; ++b; }
        const float val = acc[j][r] + bias;
        const int bh = b * 12 + head;
        if (which == 0)
          Qb[((size_t)bh * 1025 + t) * 64 + dh] = (f16)(val * 0.125f); // *1/sqrt(64)
        else if (which == 1)
          Kb[((size_t)bh * 1025 + t) * 64 + dh] = (f16)val;
        else
          Vt[((size_t)bh * 64 + dh) * 1040 + t] = (f16)val;
      }
    }
  }
}

// ===========================================================================
// Kernel 2: masked flash attention. One wave per (bh, 16-query tile).
// Only visits key rows within Chebyshev radius (+ key-step 0 for CLS col).
// ===========================================================================
__global__ __launch_bounds__(256) void attn_kernel(
    const f16* __restrict__ Qb, const f16* __restrict__ Kb,
    const f16* __restrict__ Vt, f16* __restrict__ AO)
{
  __shared__ __align__(16) f16 Pb[8][16][40];  // per-wave P transpose buffer

  const int tid = threadIdx.x, lane = tid & 31, w = tid >> 5;
  const int lm = lane & 15, hh = lane >> 4;
  const int unit = blockIdx.x * 8 + w;      // 780*8 == 96*65 exactly
  const int bh = unit / 65, qt = unit % 65;
  const int t0 = qt * 16;
  const int b = bh / 12, head = bh % 12;

  const f16* Qp = Qb + (size_t)bh * 1025 * 64;
  const f16* Kp = Kb + (size_t)bh * 1025 * 64;
  const f16* Vp = Vt + (size_t)bh * 64 * 1040;

  // Q fragments (A-matrix, dh halves d=0,1), loaded once
  V16 qf[2];
  {
    int tq = t0 + lm; if (tq > 1024) tq = 1024;
    const f16* qrow = Qp + (size_t)tq * 64;
#pragma unroll
    for (int d = 0; d < 2; ++d) {
      qf[d].h[0] = *(const v8h*)(qrow + 32 * d + 16 * hh);
      qf[d].h[1] = *(const v8h*)(qrow + 32 * d + 16 * hh + 8);
    }
  }

  v8f O[4] = {};
  float rmax[8], rsum[8];
#pragma unroll
  for (int r = 0; r < 8; ++r) { rmax[r] = NEG_INF; rsum[r] = 0.f; }

  // key-step range from mask locality (steps of 32 keys)
  int sBeg, sEnd; bool addCls;
  if (qt == 0) { sBeg = 0; sEnd = 32; addCls = false; }
  else {
    const int pmin = t0 - 1;
    int pmax = t0 + 15; if (pmax > 1024) pmax = 1024; pmax -= 1;
    int rlo = (pmin >> 5) - 3; if (rlo < 0) rlo = 0;
    int rhi = (pmax >> 5) + 3; if (rhi > 31) rhi = 31;
    sBeg = rlo; sEnd = rhi + 1; addCls = (sBeg > 0);
  }

  auto step = [&](int s) {
    const int kbase = s * 32;
    float pv[2][8];
    float tmax[8];
#pragma unroll
    for (int r = 0; r < 8; ++r) tmax[r] = NEG_INF;

    // ---- S = Q K^T for two 16-key sub-tiles ----
#pragma unroll
    for (int j = 0; j < 2; ++j) {
      const int kb = kbase + 16 * j;
      int tk = kb + lm; if (tk > 1024) tk = 1024;
      const f16* krow = Kp + (size_t)tk * 64;
      v8f sacc = {};
#pragma unroll
      for (int d = 0; d < 2; ++d) {
        V16 kf;  // B fragment: N=key(lane), K=dh run 32d+16h..+15
        kf.h[0] = *(const v8h*)(krow + 32 * d + 16 * hh);
        kf.h[1] = *(const v8h*)(krow + 32 * d + 16 * hh + 8);
        sacc = WMMA_F16(qf[d].v, kf.v, sacc);
      }
      const int tkk = kb + lm;  // unclamped key index for masking
#pragma unroll
      for (int r = 0; r < 8; ++r) {
        const int tq = t0 + r + 8 * hh;
        bool vis;
        if (tq > 1024 || tkk > 1024) vis = false;
        else if (tq == 0 || tkk == 0) vis = true;
        else {
          const int p = tq - 1, pp = tkk - 1;
          int dx = (p >> 5) - (pp >> 5); if (dx < 0) dx = -dx;
          int dy = (p & 31) - (pp & 31); if (dy < 0) dy = -dy;
          vis = (dx <= 3) && (dy <= 3);
        }
        const float sv = vis ? sacc[r] : NEG_INF;
        pv[j][r] = sv;
        tmax[r] = fmaxf(tmax[r], sv);
      }
    }

    // ---- online softmax (row reductions over 16-lane half-waves) ----
    float fac[8];
#pragma unroll
    for (int r = 0; r < 8; ++r) {
      float t = tmax[r];
      t = fmaxf(t, __shfl_xor(t, 1, 32));
      t = fmaxf(t, __shfl_xor(t, 2, 32));
      t = fmaxf(t, __shfl_xor(t, 4, 32));
      t = fmaxf(t, __shfl_xor(t, 8, 32));
      const float nm = fmaxf(rmax[r], t);
      const float m2 = (nm == NEG_INF) ? 0.f : nm;
      fac[r] = __expf(rmax[r] - m2);      // rmax=-inf -> 0 (O is 0 anyway)
      rmax[r] = nm;
      float rs = 0.f;
#pragma unroll
      for (int j = 0; j < 2; ++j) {
        const float p = __expf(pv[j][r] - m2);   // masked -inf -> 0
        pv[j][r] = p;
        rs += p;
      }
      rs += __shfl_xor(rs, 1, 32);
      rs += __shfl_xor(rs, 2, 32);
      rs += __shfl_xor(rs, 4, 32);
      rs += __shfl_xor(rs, 8, 32);
      rsum[r] = rsum[r] * fac[r] + rs;
    }
#pragma unroll
    for (int sub = 0; sub < 4; ++sub)
#pragma unroll
      for (int r = 0; r < 8; ++r) O[sub][r] *= fac[r];

    // ---- transpose P (C layout -> A layout) through per-wave LDS ----
#pragma unroll
    for (int j = 0; j < 2; ++j)
#pragma unroll
      for (int r = 0; r < 8; ++r)
        Pb[w][r + 8 * hh][16 * j + lm] = (f16)pv[j][r];
    asm volatile("s_wait_dscnt 0x0" ::: "memory");   // same-wave DS RAW
    __builtin_amdgcn_wave_barrier();
    V16 pf;
    pf.h[0] = *(const v8h*)&Pb[w][lm][16 * hh];
    pf.h[1] = *(const v8h*)&Pb[w][lm][16 * hh + 8];

    // ---- O += P @ V (4 dh sub-tiles), V read contiguously from Vt ----
    int tvb = kbase + 16 * hh; if (tvb > 1024) tvb = 1024;
#pragma unroll
    for (int sub = 0; sub < 4; ++sub) {
      const int dh = 16 * sub + lm;
      const f16* vrow = Vp + (size_t)dh * 1040 + tvb;
      V16 vf;
      vf.h[0] = *(const v8h*)(vrow);
      vf.h[1] = *(const v8h*)(vrow + 8);
      O[sub] = WMMA_F16(pf.v, vf.v, O[sub]);
    }
  };

  if (addCls) step(0);               // CLS column lives in key-step 0
  for (int s = sBeg; s <= sEnd; ++s) step(s);

  // ---- normalize and store to AO [b*1025+t][head*64+dh] (f16) ----
#pragma unroll
  for (int sub = 0; sub < 4; ++sub) {
    const int dh = 16 * sub + lm;
#pragma unroll
    for (int r = 0; r < 8; ++r) {
      const int tq = t0 + r + 8 * hh;
      if (tq <= 1024) {
        const float val = O[sub][r] / rsum[r];   // rsum>0: CLS key always visible
        AO[((size_t)b * 1025 + tq) * 768 + head * 64 + dh] = (f16)val;
      }
    }
  }
}

// ===========================================================================
// Kernel 3: out = AO @ Wpt^T + bproj  (f16 operands, f32 output)
// Same async double-buffered structure as kernel 1.
// ===========================================================================
__global__ __launch_bounds__(256) void proj_gemm_kernel(
    const f16* __restrict__ AO, const f16* __restrict__ Wt,
    const float* __restrict__ bp, float* __restrict__ out)
{
  __shared__ __align__(16) f16 As[2][64][40];
  __shared__ __align__(16) f16 Bs[2][128][40];

  const int tid  = threadIdx.x;
  const int lane = tid & 31;
  const int w    = tid >> 5;
  const int mw   = w & 3, nw = w >> 2;
  const int lm   = lane & 15, hh = lane >> 4;
  const int m0   = blockIdx.y * 64;
  const int n0   = blockIdx.x * 128;

  auto stage = [&](int kk, int buf) {
    const int k0 = kk * 32;
    {
      const int row = tid >> 2, seg = tid & 3;
      int gm = m0 + row; if (gm > 8199) gm = 8199;
      const unsigned la = (unsigned)(size_t)&As[buf][row][seg * 8];
      const int go = (gm * 768 + k0 + seg * 8) * 2;
      ASYNC_B128(la, go, AO);
    }
#pragma unroll
    for (int it = 0; it < 2; ++it) {
      const int idx = tid + it * 256;
      const int row = idx >> 2, seg = idx & 3;
      const unsigned lb = (unsigned)(size_t)&Bs[buf][row][seg * 8];
      const int go = ((n0 + row) * 768 + k0 + seg * 8) * 2;
      ASYNC_B128(lb, go, Wt);
    }
    asm volatile("s_wait_xcnt 0x0" ::: "memory");
  };

  v8f acc[4] = {};
  stage(0, 0);

  for (int kk = 0; kk < 24; ++kk) {
    const int buf = kk & 1;
    asm volatile("s_wait_asynccnt 0x0" ::: "memory");
    __syncthreads();
    if (kk + 1 < 24) stage(kk + 1, buf ^ 1);

    V16 a;
    a.h[0] = *(const v8h*)&As[buf][mw * 16 + lm][16 * hh];
    a.h[1] = *(const v8h*)&As[buf][mw * 16 + lm][16 * hh + 8];
#pragma unroll
    for (int j = 0; j < 4; ++j) {
      const f16* bptr = &Bs[buf][nw * 64 + j * 16 + lm][16 * hh];
      V16 bfr;
      bfr.h[0] = *(const v8h*)bptr;
      bfr.h[1] = *(const v8h*)(bptr + 8);
      acc[j] = WMMA_F16(a.v, bfr.v, acc[j]);
    }
    __syncthreads();
  }

#pragma unroll
  for (int j = 0; j < 4; ++j) {
    const int n = n0 + nw * 64 + j * 16 + lm;
    const float bias = bp[n];
#pragma unroll
    for (int r = 0; r < 8; ++r) {
      const int m = m0 + mw * 16 + r + 8 * hh;
      if (m < 8200) out[(size_t)m * 768 + n] = acc[j][r] + bias;
    }
  }
}

// ===========================================================================
extern "C" void kernel_launch(void* const* d_in, const int* in_sizes, int n_in,
                              void* d_out, int out_size, void* d_ws, size_t ws_size,
                              hipStream_t stream) {
  const float* x     = (const float*)d_in[0];
  const float* Wqkv  = (const float*)d_in[1];
  const float* bqkv  = (const float*)d_in[2];
  const float* Wproj = (const float*)d_in[3];
  const float* bproj = (const float*)d_in[4];
  float* out = (float*)d_out;

  char* ws = (char*)d_ws;
  f16* Qb    = (f16*)(ws);
  f16* Kb    = (f16*)(ws + 12595200);
  f16* Vt    = (f16*)(ws + 25190400);
  f16* AO    = (f16*)(ws + 37969920);   // reused: Xh before attn, AO after
  f16* Xh    = AO;
  f16* Wqkvt = (f16*)(ws + 50577408);
  f16* Wpt   = (f16*)(ws + 54116352);

  cvt_x_kernel<<<dim3(3075), dim3(256), 0, stream>>>(x, Xh);
  tconv_kernel<<<dim3(72, 24), dim3(256), 0, stream>>>(Wqkv, Wqkvt, 768, 2304);
  tconv_kernel<<<dim3(24, 24), dim3(256), 0, stream>>>(Wproj, Wpt, 768, 768);
  qkv_gemm_kernel<<<dim3(18, 129), dim3(256), 0, stream>>>(Xh, Wqkvt, bqkv, Qb, Kb, Vt);
  attn_kernel<<<dim3(780), dim3(256), 0, stream>>>(Qb, Kb, Vt, AO);
  proj_gemm_kernel<<<dim3(6, 129), dim3(256), 0, stream>>>(AO, Wpt, bproj, out);
}
